// LinearAttention_89635967468216
// MI455X (gfx1250) — matrix operand order repaired and checked by reading, not compiled, not run on hardware
//
#include <hip/hip_runtime.h>
#include <math.h>

// ---------------------------------------------------------------------------
// LinearAttention on MI455X (gfx1250, wave32, WMMA).
// Matrix math via v_wmma_f32_16x16x32_f16 (f16 in, f32 accumulate).
// Async global->LDS staging (ASYNCcnt) for the q tile in the AV kernel.
// ---------------------------------------------------------------------------

typedef _Float16 half_t;
typedef __attribute__((ext_vector_type(16))) _Float16 v16h;
typedef __attribute__((ext_vector_type(8)))  float    v8f;

#define Bz 2
#define Dm 256
#define Cc 64
#define Tt 1024
#define Hh 8
#define Ff 32
#define Ee 768              // 3*Dm
#define CT 65536            // Cc*Tt
#define Pn 131072           // Bz*CT positions
#define EPS 1e-5f

// workspace byte offsets (256-aligned)
#define MU_OFF    0ull                      // 512 f32
#define RSTD_OFF  4096ull                   // 512 f32
#define BIAS2_OFF 8192ull                   // 2*768 f32
#define WN_OFF    16384ull                  // 2*768*256 f16  (norm-folded in_w)
#define WNO_OFF   802816ull                 // 256*256 f16    (out_w as f16)
#define QKV_OFF   1048576ull                // 768*Pn f16
#define CTX_OFF   202375168ull              // 1024*32*32 f16
#define ATTN_OFF  204472320ull              // 256*Pn f16

__device__ __forceinline__ v8f v8f_zero() {
  v8f z;
#pragma unroll
  for (int i = 0; i < 8; ++i) z[i] = 0.0f;
  return z;
}

__device__ __forceinline__ v8f wmma_f16(v16h a, v16h b, v8f c) {
  return __builtin_amdgcn_wmma_f32_16x16x32_f16(false, a, false, b, (short)0, c,
                                                false, false);
}

// pack two f32 -> two f16 in one dword
__device__ __forceinline__ unsigned int pack2h(float a, float b) {
  union { half_t h[2]; unsigned int u; } t;
  t.h[0] = (half_t)a; t.h[1] = (half_t)b;
  return t.u;
}
__device__ __forceinline__ unsigned int pack2hh(half_t a, half_t b) {
  union { half_t h[2]; unsigned int u; } t;
  t.h[0] = a; t.h[1] = b;
  return t.u;
}

// ---- fragment loaders (CDNA5 documented per-lane layouts, wave32) ----------
// A 16x32 f16, element (m,k) at base[m*stride + k]  (row-major storage)
__device__ __forceinline__ v16h frag_a_rowmajor(const half_t* base, int stride,
                                                int moff, int lane) {
  int m = lane & 15, kh = lane >> 4;
  const half_t* p = base + (moff + m) * stride + kh * 8;
  v16h a;
#pragma unroll
  for (int j = 0; j < 8; ++j) a[j] = p[j];          // K = kh*8 + j
#pragma unroll
  for (int j = 0; j < 8; ++j) a[8 + j] = p[16 + j]; // K = 16 + kh*8 + j
  return a;
}
// A 16x32 f16, element (m,k) at base[k*stride + m]  (col-major storage)
__device__ __forceinline__ v16h frag_a_colmajor(const half_t* base, int stride,
                                                int moff, int lane) {
  int m = lane & 15, kh = lane >> 4;
  v16h a;
#pragma unroll
  for (int j = 0; j < 8; ++j) a[j] = base[(kh * 8 + j) * stride + moff + m];
#pragma unroll
  for (int j = 0; j < 8; ++j) a[8 + j] = base[(16 + kh * 8 + j) * stride + moff + m];
  return a;
}
// B 32x16 f16, element (k,n) at base[n*stride + k]  (N-major storage: contiguous)
__device__ __forceinline__ v16h frag_b_nmajor(const half_t* base, int stride,
                                              int noff, int lane) {
  int n = lane & 15, kh = lane >> 4;
  const half_t* p = base + (noff + n) * stride + kh * 8;
  v16h b;
#pragma unroll
  for (int j = 0; j < 8; ++j) b[j] = p[j];          // K = kh*8 + j
#pragma unroll
  for (int j = 0; j < 8; ++j) b[8 + j] = p[16 + j]; // K = 16 + kh*8 + j
  return b;
}
// B 32x16 f16, element (k,n) at base[k*stride + n]  (K-major storage: strided)
__device__ __forceinline__ v16h frag_b_kmajor(const half_t* base, int stride,
                                              int noff, int lane) {
  int n = lane & 15, kh = lane >> 4;
  v16h b;
#pragma unroll
  for (int j = 0; j < 16; ++j) b[j] = base[((j < 8 ? 0 : 16) + kh * 8 + (j & 7)) * stride + noff + n];
  return b;
}

// ---------------------------------------------------------------------------
// K1: per-(b,d) mean / rstd over 65536 spatial elements
// ---------------------------------------------------------------------------
__global__ void stats_kernel(const float* __restrict__ x,
                             float* __restrict__ mu, float* __restrict__ rstd) {
  const int bd = blockIdx.x;                  // 0..511
  const float* row = x + (size_t)bd * CT;
  float s = 0.f, ss = 0.f;
  for (int i = threadIdx.x; i < CT; i += 256) {
    float v = row[i];
    s += v; ss += v * v;
  }
  __shared__ float r0[256], r1[256];
  r0[threadIdx.x] = s; r1[threadIdx.x] = ss;
  __syncthreads();
  for (int off = 128; off > 0; off >>= 1) {
    if (threadIdx.x < off) { r0[threadIdx.x] += r0[threadIdx.x + off];
                             r1[threadIdx.x] += r1[threadIdx.x + off]; }
    __syncthreads();
  }
  if (threadIdx.x == 0) {
    float m = r0[0] * (1.0f / CT);
    float var = r1[0] * (1.0f / CT) - m * m;
    mu[bd] = m;
    rstd[bd] = rsqrtf(var + EPS);
  }
}

// ---------------------------------------------------------------------------
// K2a: fold InstanceNorm into qkv weights:  Wn[b][e][d] = w*rstd (f16),
//      bias2[b][e] = in_b[e] - sum_d w*mu*rstd
// ---------------------------------------------------------------------------
__global__ void foldw_kernel(const float* __restrict__ in_w,
                             const float* __restrict__ in_b,
                             const float* __restrict__ mu,
                             const float* __restrict__ rstd,
                             half_t* __restrict__ Wn,
                             float* __restrict__ bias2) {
  const int be = blockIdx.x;                  // 0..1535
  const int b = be / Ee, e = be % Ee;
  const int d = threadIdx.x;                  // 256 threads
  float w = in_w[e * Dm + d];
  float r = rstd[b * Dm + d];
  float m = mu[b * Dm + d];
  Wn[((size_t)b * Ee + e) * Dm + d] = (half_t)(w * r);
  __shared__ float red[256];
  red[d] = w * r * m;
  __syncthreads();
  for (int off = 128; off > 0; off >>= 1) {
    if (d < off) red[d] += red[d + off];
    __syncthreads();
  }
  if (d == 0) bias2[b * Ee + e] = in_b[e] - red[0];
}

// K2b: out_w -> f16
__global__ void convw_kernel(const float* __restrict__ out_w,
                             half_t* __restrict__ Wno) {
  int idx = blockIdx.x * 256 + threadIdx.x;   // 65536 elements
  Wno[idx] = (half_t)out_w[idx];
}

// ---------------------------------------------------------------------------
// K3: QKV GEMM  qkv[e][p] = sum_d Wn[b][e][d] * x[d][p] + bias2[b][e]
//     M=768, N=Pn, K=256.  WG tile 64x256, 8 waves (2x4), wave tile 32x64.
// ---------------------------------------------------------------------------
__global__ void qkv_gemm_kernel(const half_t* __restrict__ Wn,
                                const float* __restrict__ x,
                                const float* __restrict__ bias2,
                                half_t* __restrict__ qkv) {
  const int ptile = blockIdx.x;               // 0..511, 256 cols
  const int mbase = blockIdx.y * 64;          // 0..11
  const int b = ptile >> 8;                   // 256 ptiles per batch image
  const int plbase = (ptile & 255) * 256;     // local offset within image
  const int tid = threadIdx.x;
  const int lane = tid & 31, wid = tid >> 5;
  const int wm = wid >> 2, wn = wid & 3;

  __shared__ half_t ldsA[64 * 40];            // [m][k] row-major, stride 40
  __shared__ half_t ldsB[256 * 40];           // [n][k] col-major, stride 40

  const half_t* Ab = Wn + ((size_t)b * Ee + mbase) * Dm;
  const float*  xb = x + (size_t)b * Dm * CT;
  // per-thread prefetch offset within a k-step tile (row = tid>>3, 128B column step)
  const size_t pfoff = (size_t)(tid >> 3) * CT + plbase + (tid & 7) * 32;

  v8f acc[2][4];
#pragma unroll
  for (int i = 0; i < 2; ++i)
#pragma unroll
    for (int j = 0; j < 4; ++j) acc[i][j] = v8f_zero();

  for (int kk = 0; kk < Dm; kk += 32) {
    // prefetch next k-step of x into L2 (global_prefetch_b8); branch-free:
    // clamp to a valid row block so the address is always in-bounds.
    int pfk = kk + 32;
    if (pfk >= Dm) pfk = 0;
    __builtin_prefetch(xb + (size_t)pfk * CT + pfoff, 0, 1);

    // A tile 64x32 f16 (dword copies, coalesced)
#pragma unroll
    for (int j = 0; j < 4; ++j) {
      int idx = tid + j * 256;                // dword index, 16 dwords/row
      int r = idx >> 4;
      int ch = (idx & 15) * 2;
      *(unsigned int*)&ldsA[r * 40 + ch] =
          *(const unsigned int*)(Ab + (size_t)r * Dm + kk + ch);
    }
    // B tile 32x256: f32 x -> f16, stored N-major.
    // Unit = row-pair x col-quad: 2 x float4 global loads -> 4 packed b32 LDS stores.
#pragma unroll
    for (int j = 0; j < 4; ++j) {
      int u = tid + j * 256;                  // 1024 units
      int rp = u >> 6;                        // row-pair 0..15
      int cq = u & 63;                        // col-quad 0..63
      const float4 fa = *(const float4*)(xb + (size_t)(kk + rp * 2) * CT + plbase + cq * 4);
      const float4 fb = *(const float4*)(xb + (size_t)(kk + rp * 2 + 1) * CT + plbase + cq * 4);
      *(unsigned int*)&ldsB[(cq * 4 + 0) * 40 + rp * 2] = pack2h(fa.x, fb.x);
      *(unsigned int*)&ldsB[(cq * 4 + 1) * 40 + rp * 2] = pack2h(fa.y, fb.y);
      *(unsigned int*)&ldsB[(cq * 4 + 2) * 40 + rp * 2] = pack2h(fa.z, fb.z);
      *(unsigned int*)&ldsB[(cq * 4 + 3) * 40 + rp * 2] = pack2h(fa.w, fb.w);
    }
    __syncthreads();

    v16h afrag[2], bfrag[4];
#pragma unroll
    for (int mt = 0; mt < 2; ++mt)
      afrag[mt] = frag_a_rowmajor(ldsA, 40, wm * 32 + mt * 16, lane);
#pragma unroll
    for (int nt = 0; nt < 4; ++nt)
      bfrag[nt] = frag_b_nmajor(ldsB, 40, wn * 64 + nt * 16, lane);
#pragma unroll
    for (int mt = 0; mt < 2; ++mt)
#pragma unroll
      for (int nt = 0; nt < 4; ++nt)
        acc[mt][nt] = wmma_f16(afrag[mt], bfrag[nt], acc[mt][nt]);
    __syncthreads();
  }

  // epilogue: +bias2, store f16
  const int mh = lane >> 4, nn = lane & 15;
#pragma unroll
  for (int mt = 0; mt < 2; ++mt)
#pragma unroll
    for (int nt = 0; nt < 4; ++nt)
#pragma unroll
      for (int r = 0; r < 8; ++r) {
        int e = mbase + wm * 32 + mt * 16 + mh * 8 + r;
        int p = ptile * 256 + wn * 64 + nt * 16 + nn;
        qkv[(size_t)e * Pn + p] = (half_t)(acc[mt][nt][r] + bias2[b * Ee + e]);
      }
}

// ---------------------------------------------------------------------------
// K4: k-softmax over t (contiguous 1024-segments), in-place on qkv rows 256..511
// ---------------------------------------------------------------------------
__global__ void ksoftmax_kernel(half_t* __restrict__ qkv) {
  const int rowid = blockIdx.x;               // 0..32767
  const int ek = rowid >> 7;                  // 0..255
  const int n = rowid & 127;
  half_t* row = qkv + (size_t)(Dm + ek) * Pn + (size_t)n * Tt;
  const int tid = threadIdx.x;
  float v[4];
  float mx = -1e30f;
#pragma unroll
  for (int j = 0; j < 4; ++j) { v[j] = (float)row[tid + j * 256]; mx = fmaxf(mx, v[j]); }
  __shared__ float red[256];
  red[tid] = mx; __syncthreads();
  for (int off = 128; off > 0; off >>= 1) {
    if (tid < off) red[tid] = fmaxf(red[tid], red[tid + off]);
    __syncthreads();
  }
  mx = red[0]; __syncthreads();
  float s = 0.f;
#pragma unroll
  for (int j = 0; j < 4; ++j) { v[j] = __expf(v[j] - mx); s += v[j]; }
  red[tid] = s; __syncthreads();
  for (int off = 128; off > 0; off >>= 1) {
    if (tid < off) red[tid] += red[tid + off];
    __syncthreads();
  }
  float inv = 1.0f / red[0];
#pragma unroll
  for (int j = 0; j < 4; ++j) row[tid + j * 256] = (half_t)(v[j] * inv);
}

// ---------------------------------------------------------------------------
// K5: q-softmax over feature dim (32, stride Pn), *1/sqrt(32), in-place rows 0..255
// ---------------------------------------------------------------------------
__global__ void qsoftmax_kernel(half_t* __restrict__ qkv) {
  const int p = blockIdx.x * 256 + threadIdx.x;   // gridDim.x = 512
  const int h = blockIdx.y;
  half_t* base = qkv + (size_t)(h * Ff) * Pn + p;
  float v[32];
  float mx = -1e30f;
#pragma unroll
  for (int j = 0; j < 32; ++j) { v[j] = (float)base[(size_t)j * Pn]; mx = fmaxf(mx, v[j]); }
  float s = 0.f;
#pragma unroll
  for (int j = 0; j < 32; ++j) { v[j] = __expf(v[j] - mx); s += v[j]; }
  float sc = 0.17677669529663689f / s;            // (1/sqrt(32)) / sum
#pragma unroll
  for (int j = 0; j < 32; ++j) base[(size_t)j * Pn] = (half_t)(v[j] * sc);
}

// ---------------------------------------------------------------------------
// K6: ctx[nh][kf][vf] = (1/T) * sum_t k[kf][t] * v[vf][t]   (32x32, K=1024)
//     one wave per (n,h); fragments straight from global (contiguous b128s).
// ---------------------------------------------------------------------------
__global__ void ctx_kernel(const half_t* __restrict__ qkv,
                           half_t* __restrict__ ctx) {
  const int wid = threadIdx.x >> 5, lane = threadIdx.x & 31;
  const int nh = blockIdx.x * 4 + wid;        // 0..1023
  const int n = nh >> 3, h = nh & 7;
  const half_t* krows = qkv + (size_t)(Dm + h * Ff) * Pn + (size_t)n * Tt;
  const half_t* vrows = qkv + (size_t)(2 * Dm + h * Ff) * Pn + (size_t)n * Tt;

  v8f acc[2][2];
#pragma unroll
  for (int i = 0; i < 2; ++i)
#pragma unroll
    for (int j = 0; j < 2; ++j) acc[i][j] = v8f_zero();

  for (int kk = 0; kk < Tt; kk += 32) {
    v16h a0 = frag_a_rowmajor(krows + kk, Pn, 0, lane);
    v16h a1 = frag_a_rowmajor(krows + kk, Pn, 16, lane);
    v16h b0 = frag_b_nmajor(vrows + kk, Pn, 0, lane);
    v16h b1 = frag_b_nmajor(vrows + kk, Pn, 16, lane);
    acc[0][0] = wmma_f16(a0, b0, acc[0][0]);
    acc[0][1] = wmma_f16(a0, b1, acc[0][1]);
    acc[1][0] = wmma_f16(a1, b0, acc[1][0]);
    acc[1][1] = wmma_f16(a1, b1, acc[1][1]);
  }
  const int mh = lane >> 4, nn = lane & 15;
  const float invT = 1.0f / (float)Tt;        // fold v/T here
#pragma unroll
  for (int mt = 0; mt < 2; ++mt)
#pragma unroll
    for (int nt = 0; nt < 2; ++nt)
#pragma unroll
      for (int r = 0; r < 8; ++r) {
        int m = mt * 16 + mh * 8 + r;         // kf
        int c = nt * 16 + nn;                 // vf
        ctx[(size_t)nh * 1024 + m * 32 + c] = (half_t)(acc[mt][nt][r] * invT);
      }
}

// ---------------------------------------------------------------------------
// K7: attn[vf][t] = sum_kf ctx[kf][vf] * q[kf][t]   (M=32, N=1024, K=32)
//     WG handles one (n,h) x half-T; q tile staged in LDS with
//     global_load_async_to_lds_b128 (ASYNCcnt path).
// ---------------------------------------------------------------------------
__global__ void av_kernel(const half_t* __restrict__ ctx,
                          const half_t* __restrict__ qkv,
                          half_t* __restrict__ attn) {
  const int nh = blockIdx.x >> 1;             // 0..1023
  const int thalf = (blockIdx.x & 1) * 512;
  const int n = nh >> 3, h = nh & 7;
  const int tid = threadIdx.x;                // 128 threads, 4 waves
  const int lane = tid & 31, wid = tid >> 5;

  __shared__ half_t ldsQ[32 * 528];           // [kf][t] row-major, stride 528

  // wave-relative LDS byte offset of ldsQ (low 32 bits of the flat address)
  const unsigned ldsq_base = (unsigned)(size_t)(void*)&ldsQ[0];
  const half_t* qbase = qkv + (size_t)(h * Ff) * Pn + (size_t)n * Tt + thalf;

  // async copy 16 x 1024 f16 tile, one b128 per lane-unit
#pragma unroll
  for (int j = 0; j < 16; ++j) {
    int u = tid + j * 128;                    // 2048 units of 8 halves
    int k = u >> 7;                           // 128 units per row of 1024
    int cu = u & 127;
    unsigned long long ga = (unsigned long long)(const void*)(qbase + (size_t)k * Pn + cu * 8);
    unsigned la = ldsq_base + (unsigned)(k * 528 + cu * 8) * 2u;
    asm volatile("global_load_async_to_lds_b128 %0, %1, off"
                 :: "v"(la), "v"(ga) : "memory");
  }
  asm volatile("s_wait_asynccnt 0x0" ::: "memory");
  __syncthreads();

  const half_t* ctxp = ctx + (size_t)nh * 1024;
  v16h a0 = frag_a_colmajor(ctxp, 32, 0, lane);   // A = ctx^T
  v16h a1 = frag_a_colmajor(ctxp, 32, 16, lane);

  const int mh = lane >> 4, nn = lane & 15;
  const int tbase = wid * 128;                // each wave owns 128 t
#pragma unroll
  for (int nt = 0; nt < 8; ++nt) {
    v16h bq = frag_b_kmajor(ldsQ, 528, tbase + nt * 16, lane);
    v8f d0 = wmma_f16(a0, bq, v8f_zero());
    v8f d1 = wmma_f16(a1, bq, v8f_zero());
    int p = n * Tt + thalf + tbase + nt * 16 + nn;
#pragma unroll
    for (int r = 0; r < 8; ++r) {
      int e0 = h * Ff + mh * 8 + r;           // vf channel, m-tile 0
      int e1 = e0 + 16;                       // m-tile 1
      attn[(size_t)e0 * Pn + p] = (half_t)d0[r];
      attn[(size_t)e1 * Pn + p] = (half_t)d1[r];
    }
  }
}

// ---------------------------------------------------------------------------
// K8: out projection  out[e][p] = sum_d Wno[e][d]*attn[d][p] + out_b[e] + x[e][p]
//     M=256, N=Pn, K=256. Same tiling as K3, f32 output with residual.
// ---------------------------------------------------------------------------
__global__ void out_gemm_kernel(const half_t* __restrict__ Wno,
                                const half_t* __restrict__ attn,
                                const float* __restrict__ out_b,
                                const float* __restrict__ x,
                                float* __restrict__ out) {
  const int ptile = blockIdx.x;               // 0..511
  const int mbase = blockIdx.y * 64;          // 0..3
  const int tid = threadIdx.x;
  const int lane = tid & 31, wid = tid >> 5;
  const int wm = wid >> 2, wn = wid & 3;

  __shared__ half_t ldsA[64 * 40];
  __shared__ half_t ldsB[256 * 40];

  const half_t* Ab = Wno + (size_t)mbase * Dm;

  v8f acc[2][4];
#pragma unroll
  for (int i = 0; i < 2; ++i)
#pragma unroll
    for (int j = 0; j < 4; ++j) acc[i][j] = v8f_zero();

  for (int kk = 0; kk < Dm; kk += 32) {
#pragma unroll
    for (int j = 0; j < 4; ++j) {
      int idx = tid + j * 256;
      int r = idx >> 4;
      int ch = (idx & 15) * 2;
      *(unsigned int*)&ldsA[r * 40 + ch] =
          *(const unsigned int*)(Ab + (size_t)r * Dm + kk + ch);
    }
    // B tile 32x256 f16: unit = row-pair x col-quad, b64 loads + packed b32 stores
#pragma unroll
    for (int j = 0; j < 4; ++j) {
      int u = tid + j * 256;
      int rp = u >> 6;
      int cq = u & 63;
      union { half_t h[4]; uint2 u2; } ra, rb;
      ra.u2 = *(const uint2*)(attn + (size_t)(kk + rp * 2) * Pn + ptile * 256 + cq * 4);
      rb.u2 = *(const uint2*)(attn + (size_t)(kk + rp * 2 + 1) * Pn + ptile * 256 + cq * 4);
#pragma unroll
      for (int i = 0; i < 4; ++i)
        *(unsigned int*)&ldsB[(cq * 4 + i) * 40 + rp * 2] = pack2hh(ra.h[i], rb.h[i]);
    }
    __syncthreads();

    v16h afrag[2], bfrag[4];
#pragma unroll
    for (int mt = 0; mt < 2; ++mt)
      afrag[mt] = frag_a_rowmajor(ldsA, 40, wm * 32 + mt * 16, lane);
#pragma unroll
    for (int nt = 0; nt < 4; ++nt)
      bfrag[nt] = frag_b_nmajor(ldsB, 40, wn * 64 + nt * 16, lane);
#pragma unroll
    for (int mt = 0; mt < 2; ++mt)
#pragma unroll
      for (int nt = 0; nt < 4; ++nt)
        acc[mt][nt] = wmma_f16(afrag[mt], bfrag[nt], acc[mt][nt]);
    __syncthreads();
  }

  const int mh = lane >> 4, nn = lane & 15;
#pragma unroll
  for (int mt = 0; mt < 2; ++mt)
#pragma unroll
    for (int nt = 0; nt < 4; ++nt)
#pragma unroll
      for (int r = 0; r < 8; ++r) {
        int e = mbase + wm * 32 + mt * 16 + mh * 8 + r;
        int p = ptile * 256 + wn * 64 + nt * 16 + nn;
        int b = p >> 16;
        int pl = p & (CT - 1);
        size_t off = ((size_t)(b * Dm + e)) * CT + pl;
        out[off] = acc[mt][nt][r] + out_b[e] + x[off];
      }
}

// ---------------------------------------------------------------------------
extern "C" void kernel_launch(void* const* d_in, const int* in_sizes, int n_in,
                              void* d_out, int out_size, void* d_ws, size_t ws_size,
                              hipStream_t stream) {
  (void)in_sizes; (void)n_in; (void)out_size; (void)ws_size;
  const float* x     = (const float*)d_in[0];
  const float* in_w  = (const float*)d_in[1];
  const float* in_b  = (const float*)d_in[2];
  const float* out_w = (const float*)d_in[3];
  const float* out_b = (const float*)d_in[4];
  float* out = (float*)d_out;

  char* ws = (char*)d_ws;                     // requires ~272 MB scratch
  float*  mu    = (float*)(ws + MU_OFF);
  float*  rstd  = (float*)(ws + RSTD_OFF);
  float*  bias2 = (float*)(ws + BIAS2_OFF);
  half_t* Wn    = (half_t*)(ws + WN_OFF);
  half_t* Wno   = (half_t*)(ws + WNO_OFF);
  half_t* qkv   = (half_t*)(ws + QKV_OFF);
  half_t* ctx   = (half_t*)(ws + CTX_OFF);
  half_t* attn  = (half_t*)(ws + ATTN_OFF);

  stats_kernel   <<<512,           256, 0, stream>>>(x, mu, rstd);
  foldw_kernel   <<<Bz * Ee,       256, 0, stream>>>(in_w, in_b, mu, rstd, Wn, bias2);
  convw_kernel   <<<256,           256, 0, stream>>>(out_w, Wno);
  qkv_gemm_kernel<<<dim3(512, 12), 256, 0, stream>>>(Wn, x, bias2, qkv);
  ksoftmax_kernel<<<32768,         256, 0, stream>>>(qkv);
  qsoftmax_kernel<<<dim3(512, 8),  256, 0, stream>>>(qkv);
  ctx_kernel     <<<256,           128, 0, stream>>>(qkv, ctx);
  av_kernel      <<<2048,          128, 0, stream>>>(ctx, qkv, attn);
  out_gemm_kernel<<<dim3(512, 4),  256, 0, stream>>>(Wno, attn, out_b, x, out);
}
